// BiLSTMEncoder_87771951661096
// MI455X (gfx1250) — matrix operand order repaired and checked by reading, not compiled, not run on hardware
//
#include <hip/hip_runtime.h>

typedef __attribute__((ext_vector_type(16))) _Float16 v16h;
typedef __attribute__((ext_vector_type(8)))  float    v8f;
typedef __attribute__((ext_vector_type(8)))  unsigned int v8u;

constexpr int B_ = 128;   // batch
constexpr int L_ = 512;   // sequence length
constexpr int D_ = 256;   // input dim
constexpr int H_ = 128;   // hidden
constexpr int G_ = 512;   // 4*H gates

// ---- workspace layout (bytes) ----
constexpr size_t OFF_XH   = 0;                                  // f16 x   [B][L][D]
constexpr size_t OFF_WCF  = OFF_XH  + (size_t)B_*L_*D_*2;       // f16 WcF [G][D]
constexpr size_t OFF_WCB  = OFF_WCF + (size_t)G_*D_*2;          // f16 WcB [G][D]
constexpr size_t OFF_BCF  = OFF_WCB + (size_t)G_*D_*2;          // f32 bcF [G]
constexpr size_t OFF_BCB  = OFF_BCF + (size_t)G_*4;             // f32 bcB [G]
constexpr size_t OFF_PREF = OFF_BCB + (size_t)G_*4;             // f32 preF [L][B][G]
constexpr size_t OFF_PREB = OFF_PREF + (size_t)L_*B_*G_*4;      // f32 preB [L][B][G]
constexpr size_t OFF_HBUF = OFF_PREB + (size_t)L_*B_*G_*4;      // f32 hbuf [B][L][2H]

__device__ __forceinline__ float sigf(float x) {
    return 1.0f / (1.0f + __expf(-x));
}
__device__ __forceinline__ float tanh_fast(float x) {
    float e = __expf(2.0f * x);
    return (e - 1.0f) / (e + 1.0f);
}

// ---------------------------------------------------------------------------
// 1) x (f32) -> xh (f16), same [B][L][D] layout
// ---------------------------------------------------------------------------
__global__ void convert_x_kernel(const float* __restrict__ x,
                                 _Float16* __restrict__ xh, int n) {
    int i = blockIdx.x * blockDim.x + threadIdx.x;
    if (i < n) xh[i] = (_Float16)x[i];
}

// ---------------------------------------------------------------------------
// 2) Fold projection into input weights:
//    Wc[g][d] = sum_e Wih[g][e] * Wproj[e][d]   (f16 out)
//    bc[g]    = bias[g] + sum_e Wih[g][e] * bproj[e]
// ---------------------------------------------------------------------------
__global__ void fold_kernel(const float* __restrict__ Wih,
                            const float* __restrict__ Wproj,
                            const float* __restrict__ bproj,
                            const float* __restrict__ bias,
                            _Float16* __restrict__ Wc,
                            float* __restrict__ bc) {
    int i = blockIdx.x * blockDim.x + threadIdx.x;
    if (i >= G_ * D_) return;
    int g = i / D_, d = i % D_;
    float acc = 0.f;
    for (int e = 0; e < D_; ++e) acc += Wih[g * D_ + e] * Wproj[e * D_ + d];
    Wc[i] = (_Float16)acc;
    if (d == 0) {
        float bb = bias[g];
        for (int e = 0; e < D_; ++e) bb += Wih[g * D_ + e] * bproj[e];
        bc[g] = bb;
    }
}

// ---------------------------------------------------------------------------
// 3) Gate-input GEMM: pre[l][b][g] = xh[b][lx][:] . Wc[g][:] + bc[g]
//    Each wave computes a 16x64 strip (4 adjacent 16x16 C tiles sharing A),
//    K = 256 = 8 chunks of k32.
// ---------------------------------------------------------------------------
__global__ void __launch_bounds__(256) pre_gemm_kernel(
        const _Float16* __restrict__ xh,
        const _Float16* __restrict__ WcF, const _Float16* __restrict__ WcB,
        const float* __restrict__ bcF, const float* __restrict__ bcB,
        float* __restrict__ preF, float* __restrict__ preB) {
    const int lane = threadIdx.x & 31;
    const int wave = threadIdx.x >> 5;
    // jobs: 2 dirs * 4096 row tiles * 8 col groups (of 64 cols) = 65536
    int job = blockIdx.x * 8 + wave;
    int dir = job >> 15;
    int jr  = job & 32767;
    int rowT = jr >> 3;             // 0..4095
    int cg   = jr & 7;              // col group: g in [cg*64, cg*64+64)
    int l  = rowT >> 3;             // time step
    int b0 = (rowT & 7) << 4;       // batch tile start
    int lx = dir ? (L_ - 1 - l) : l;
    const _Float16* Wc  = dir ? WcB : WcF;
    const float*    bc  = dir ? bcB : bcF;
    float*          pre = dir ? preB : preF;

    const int ln = lane & 15;
    const int hi = lane >> 4;

    v8f acc[4];
#pragma unroll
    for (int j = 0; j < 4; ++j) {
        float bcv = bc[cg * 64 + j * 16 + ln];
#pragma unroll
        for (int r = 0; r < 8; ++r) acc[j][r] = bcv;
    }

    const size_t aBase = ((size_t)(b0 + ln) * L_ + lx) * D_;  // x[b][lx][*]

#pragma unroll
    for (int ch = 0; ch < 8; ++ch) {
        const int koff = ch << 5;
        v8u au;
#pragma unroll
        for (int v = 0; v < 8; ++v) {
            int ka = koff + 2 * v + (v >= 4 ? 8 : 0) + (hi ? 8 : 0);  // A layout
            au[v] = *(const unsigned int*)(xh + aBase + ka);
        }
        v16h a = __builtin_bit_cast(v16h, au);
#pragma unroll
        for (int j = 0; j < 4; ++j) {
            const size_t bBase = (size_t)(cg * 64 + j * 16 + ln) * D_;
            v8u bu;
#pragma unroll
            for (int v = 0; v < 8; ++v) {
                int kb = koff + 2 * v + (v >= 4 ? 8 : 0) + (hi ? 8 : 0);  // B layout
                bu[v] = *(const unsigned int*)(Wc + bBase + kb);
            }
            v16h b = __builtin_bit_cast(v16h, bu);
            acc[j] = __builtin_amdgcn_wmma_f32_16x16x32_f16(false, a, false, b,
                                                            (short)0, acc[j], false, false);
        }
    }
    // store C tiles: row = b0 + r + 8*hi, col = cg*64 + j*16 + ln
#pragma unroll
    for (int j = 0; j < 4; ++j) {
        int g = cg * 64 + j * 16 + ln;
#pragma unroll
        for (int r = 0; r < 8; ++r) {
            int b = b0 + r + (hi ? 8 : 0);
            pre[((size_t)l * B_ + b) * G_ + g] = acc[j][r];
        }
    }
}

// ---------------------------------------------------------------------------
// 4) LSTM scan. Grid = 16 workgroups: blockIdx = dir*8 + batchSlice.
//    Each workgroup: 16 batch rows, 256 threads (8 waves), Whh resident in
//    LDS (f16, 128 KB), cell state in VGPRs, h double-buffered in 2x4 KB LDS
//    (one barrier per step), pre[t+1] prefetched into registers during step t.
//    Per step per wave: 16 x v_wmma_f32_16x16x32_f16 (4 gates x K=128).
// ---------------------------------------------------------------------------
__global__ void __launch_bounds__(256) lstm_scan_kernel(
        const float* __restrict__ WhhF, const float* __restrict__ WhhB,
        const float* __restrict__ preF, const float* __restrict__ preB,
        float* __restrict__ hbuf) {
    extern __shared__ char smem_raw[];
    _Float16* whh  = (_Float16*)smem_raw;                            // [G_][H_]
    _Float16* hsh  = (_Float16*)(smem_raw + (size_t)G_ * H_ * 2);    // [2][16*H_]

    const int dir = blockIdx.x >> 3;
    const int m0  = (blockIdx.x & 7) << 4;   // batch slice start
    const float* Whh = dir ? WhhB : WhhF;
    const float* pre = dir ? preB : preF;
    const int tid = threadIdx.x;

    for (int i = tid; i < G_ * H_; i += 256) whh[i] = (_Float16)Whh[i];
    for (int i = tid; i < 2 * 16 * H_; i += 256) hsh[i] = (_Float16)0.f;
    __syncthreads();

    const int lane = tid & 31;
    const int ln   = lane & 15;
    const int hi   = lane >> 4;
    const int n0   = (tid >> 5) << 4;        // hidden tile per wave (0..112)
    const int rowOff = m0 + (hi ? 8 : 0);    // batch row base for C layout

    v8f cv;
#pragma unroll
    for (int r = 0; r < 8; ++r) cv[r] = 0.f;

    // prefetch pre for t = 0
    v8f preReg[4];
#pragma unroll
    for (int q = 0; q < 4; ++q) {
        const int gcol = q * H_ + n0 + ln;
#pragma unroll
        for (int r = 0; r < 8; ++r)
            preReg[q][r] = pre[((size_t)0 * B_ + rowOff + r) * G_ + gcol];
    }

#pragma unroll 2
    for (int t = 0; t < L_; ++t) {
        // ---- issue prefetch of pre[t+1] (off the critical path) ----
        v8f preNext[4];
        const int tn = (t + 1 < L_) ? (t + 1) : t;
#pragma unroll
        for (int q = 0; q < 4; ++q) {
            const int gcol = q * H_ + n0 + ln;
#pragma unroll
            for (int r = 0; r < 8; ++r)
                preNext[q][r] = pre[((size_t)tn * B_ + rowOff + r) * G_ + gcol];
        }

        // ---- A tiles: h slice [16 x 128] from LDS read buffer (t&1) ----
        const _Float16* hrd = hsh + (size_t)(t & 1) * 16 * H_;
        v16h A[4];
#pragma unroll
        for (int ck = 0; ck < 4; ++ck) {
            v8u au;
#pragma unroll
            for (int v = 0; v < 8; ++v) {
                int k = (ck << 5) + 2 * v + (v >= 4 ? 8 : 0) + (hi ? 8 : 0);
                au[v] = *(const unsigned int*)(hrd + ln * H_ + k);
            }
            A[ck] = __builtin_bit_cast(v16h, au);
        }

        // ---- 4 gate tiles: C = pre[t] + h @ Whh^T ----
        v8f gate[4];
#pragma unroll
        for (int q = 0; q < 4; ++q) {
            const int gcol = q * H_ + n0 + ln;
            v8f acc = preReg[q];
#pragma unroll
            for (int ck = 0; ck < 4; ++ck) {
                v8u bu;
#pragma unroll
                for (int v = 0; v < 8; ++v) {
                    int k = (ck << 5) + 2 * v + (v >= 4 ? 8 : 0) + (hi ? 8 : 0);
                    bu[v] = *(const unsigned int*)(whh + gcol * H_ + k);
                }
                v16h bm = __builtin_bit_cast(v16h, bu);
                acc = __builtin_amdgcn_wmma_f32_16x16x32_f16(false, A[ck], false, bm,
                                                             (short)0, acc, false, false);
            }
            gate[q] = acc;
        }

        // ---- gate nonlinearities, state update (PyTorch order i,f,g,o) ----
        v8f h8;
#pragma unroll
        for (int r = 0; r < 8; ++r) {
            float cs = sigf(gate[1][r]) * cv[r] + sigf(gate[0][r]) * tanh_fast(gate[2][r]);
            float hh = sigf(gate[3][r]) * tanh_fast(cs);
            cv[r] = cs;
            h8[r] = hh;
        }

        // ---- global store h into [B][L][2H] ----
        const int lOut = dir ? (L_ - 1 - t) : t;
        const int col  = dir * H_ + n0 + ln;
#pragma unroll
        for (int r = 0; r < 8; ++r) {
            hbuf[((size_t)(rowOff + r) * L_ + lOut) * (2 * H_) + col] = h8[r];
        }

        // ---- write new h into LDS write buffer ((t+1)&1) ----
        _Float16* hwr = hsh + (size_t)((t + 1) & 1) * 16 * H_;
#pragma unroll
        for (int r = 0; r < 8; ++r) {
            int rowLoc = r + (hi ? 8 : 0);
            hwr[rowLoc * H_ + n0 + ln] = (_Float16)h8[r];
        }
        __syncthreads();   // single barrier: writes visible before next read

        // rotate prefetch buffer
#pragma unroll
        for (int q = 0; q < 4; ++q) preReg[q] = preNext[q];
    }
}

// ---------------------------------------------------------------------------
// 5) LayerNorm over last dim (256). One wave per row, wave32 shuffle reduce.
// ---------------------------------------------------------------------------
__global__ void __launch_bounds__(256) ln_kernel(
        const float* __restrict__ hbuf,
        const float* __restrict__ gamma, const float* __restrict__ beta,
        float* __restrict__ out) {
    const int lane = threadIdx.x & 31;
    const int wave = threadIdx.x >> 5;
    const size_t row = (size_t)blockIdx.x * 8 + wave;
    const float* src = hbuf + row * (2 * H_);
    float v[8], s = 0.f, s2 = 0.f;
#pragma unroll
    for (int j = 0; j < 8; ++j) {
        v[j] = src[lane + 32 * j];
        s  += v[j];
        s2 += v[j] * v[j];
    }
#pragma unroll
    for (int m = 16; m >= 1; m >>= 1) {
        s  += __shfl_xor(s,  m, 32);
        s2 += __shfl_xor(s2, m, 32);
    }
    const float mu  = s * (1.0f / 256.0f);
    const float var = s2 * (1.0f / 256.0f) - mu * mu;
    const float inv = rsqrtf(var + 1e-5f);
    float* dst = out + row * (2 * H_);
#pragma unroll
    for (int j = 0; j < 8; ++j) {
        int c = lane + 32 * j;
        dst[c] = (v[j] - mu) * inv * gamma[c] + beta[c];
    }
}

// ---------------------------------------------------------------------------
extern "C" void kernel_launch(void* const* d_in, const int* in_sizes, int n_in,
                              void* d_out, int out_size, void* d_ws, size_t ws_size,
                              hipStream_t stream) {
    (void)in_sizes; (void)n_in; (void)out_size; (void)ws_size;
    const float* x     = (const float*)d_in[0];
    const float* Wproj = (const float*)d_in[1];
    const float* bproj = (const float*)d_in[2];
    const float* WihF  = (const float*)d_in[3];
    const float* WhhF  = (const float*)d_in[4];
    const float* bF    = (const float*)d_in[5];
    const float* WihB  = (const float*)d_in[6];
    const float* WhhB  = (const float*)d_in[7];
    const float* bB    = (const float*)d_in[8];
    const float* gamma = (const float*)d_in[9];
    const float* beta  = (const float*)d_in[10];
    float* out = (float*)d_out;

    char* ws = (char*)d_ws;
    _Float16* xh   = (_Float16*)(ws + OFF_XH);
    _Float16* WcF  = (_Float16*)(ws + OFF_WCF);
    _Float16* WcB  = (_Float16*)(ws + OFF_WCB);
    float*    bcF  = (float*)(ws + OFF_BCF);
    float*    bcB  = (float*)(ws + OFF_BCB);
    float*    preF = (float*)(ws + OFF_PREF);
    float*    preB = (float*)(ws + OFF_PREB);
    float*    hbuf = (float*)(ws + OFF_HBUF);

    const int nx = B_ * L_ * D_;
    convert_x_kernel<<<(nx + 255) / 256, 256, 0, stream>>>(x, xh, nx);

    fold_kernel<<<(G_ * D_ + 255) / 256, 256, 0, stream>>>(WihF, Wproj, bproj, bF, WcF, bcF);
    fold_kernel<<<(G_ * D_ + 255) / 256, 256, 0, stream>>>(WihB, Wproj, bproj, bB, WcB, bcB);

    // 2 dirs * 4096 row tiles * 8 col groups = 65536 wave jobs, 8 per block
    pre_gemm_kernel<<<8192, 256, 0, stream>>>(xh, WcF, WcB, bcF, bcB, preF, preB);

    const size_t smem = (size_t)G_ * H_ * 2 + (size_t)2 * 16 * H_ * 2;  // 139264 B
    lstm_scan_kernel<<<16, 256, smem, stream>>>(WhhF, WhhB, preF, preB, hbuf);

    ln_kernel<<<(B_ * L_) / 8, 256, 0, stream>>>(hbuf, gamma, beta, out);
}
